// GPT_50165218017588
// MI455X (gfx1250) — matrix operand order repaired and checked by reading, not compiled, not run on hardware
//
#include <hip/hip_runtime.h>

// ---------------- constants (match reference) ----------------
#define LNUM 8
#define HN   16
#define DIM  1024
#define HD   64
#define RNK  64
#define FF   4096
#define VOC  50257
#define SEQ  1024
#define BAT  2
#define MROWS (BAT*SEQ)   // 2048

typedef __bf16 bf16;
typedef __attribute__((ext_vector_type(16))) __bf16 v16bf;
typedef __attribute__((ext_vector_type(8)))  float  v8f;
typedef __attribute__((ext_vector_type(4)))  unsigned int u32x4;

// ---------------- WMMA helpers ----------------
// A-fragment (16x32 bf16, MxK): lanes 0-15 hold M=lane, K=0..15;
// lanes 16-31 hold M=lane-16, K=16..31.  seg points at the lane's
// 32-element K segment (row-major, K contiguous).
__device__ __forceinline__ v16bf load_a_frag(const bf16* seg) {
  unsigned lane = threadIdx.x & 31u;
  unsigned hi   = lane >> 4;
  union { u32x4 q[2]; v16bf v; } t;
  t.q[0] = *(const u32x4*)(seg + (hi ? 16 : 0));
  t.q[1] = *(const u32x4*)(seg + (hi ? 24 : 8));
  return t.v;
}

// B-fragment (32x16 bf16, KxN) stored as [N][K] (K contiguous per column):
// lanes 0-15: N=lane, K=0..15 ; lanes 16-31: N=lane-16, K=16..31.
__device__ __forceinline__ v16bf load_b_frag(const bf16* seg) {
  unsigned lane = threadIdx.x & 31u;
  return *(const v16bf*)(seg + ((lane >> 4) ? 16 : 0));
}

__device__ __forceinline__ v8f wmma_bf16(v16bf a, v16bf b, v8f c) {
  return __builtin_amdgcn_wmma_f32_16x16x32_bf16(false, a, false, b, (short)0, c,
                                                 false, false);
}

__device__ __forceinline__ float gelu_f(float x) {
  const float c = 0.7978845608028654f;  // sqrt(2/pi)
  float t = tanhf(c * (x + 0.044715f * x * x * x));
  return 0.5f * x * (1.0f + t);
}

// ---------------- embed: x = wte[idx] + wpe ----------------
__global__ __launch_bounds__(256) void embed_kernel(const int* __restrict__ idx,
                                                    const float* __restrict__ wte,
                                                    const float* __restrict__ wpe,
                                                    float* __restrict__ x) {
  int row = blockIdx.x;            // b*SEQ + s
  int s   = row & (SEQ - 1);
  int tok = idx[row];
  const float* we = wte + (size_t)tok * DIM;
  const float* wp = wpe + (size_t)s   * DIM;
  float* xr = x + (size_t)row * DIM;
  for (int i = threadIdx.x; i < DIM; i += 256) xr[i] = we[i] + wp[i];
}

// ---------------- layernorm -> bf16 ----------------
__global__ __launch_bounds__(256) void ln_kernel(const float* __restrict__ xin,
                                                 const float* __restrict__ g,
                                                 const float* __restrict__ bta,
                                                 bf16* __restrict__ outb) {
  __shared__ float red[256];
  int row = blockIdx.x, tid = threadIdx.x;
  const float* xr = xin + (size_t)row * DIM;
  float s = 0.f;
  for (int i = tid; i < DIM; i += 256) s += xr[i];
  red[tid] = s; __syncthreads();
  for (int off = 128; off > 0; off >>= 1) {
    if (tid < off) red[tid] += red[tid + off];
    __syncthreads();
  }
  float mu = red[0] * (1.f / DIM);
  __syncthreads();
  float v = 0.f;
  for (int i = tid; i < DIM; i += 256) { float d = xr[i] - mu; v += d * d; }
  red[tid] = v; __syncthreads();
  for (int off = 128; off > 0; off >>= 1) {
    if (tid < off) red[tid] += red[tid + off];
    __syncthreads();
  }
  float rstd = rsqrtf(red[0] * (1.f / DIM) + 1e-5f);
  for (int i = tid; i < DIM; i += 256)
    outb[(size_t)row * DIM + i] = (bf16)((xr[i] - mu) * rstd * g[i] + bta[i]);
}

// ---------------- generic bf16 WMMA GEMM (double-buffered LDS pipeline) ---
// C[M,N] = act( A_bf16[M,K] @ Bw_f32 + bias (+ residual) )
// Weight addressing handles per-head layouts:
//   B(k, n) = Bw[ (n/head_cols)*head_stride + k*ldb + (n%head_cols) ]
// Block = 128 thr (4 waves), tile 128x128, each wave 64x64 (4x4 WMMA / kstep).
// Per k-step: stage converted bf16 tile with 4x ds_store_b128/thread, one
// barrier, issue next tile's global loads, then 16 WMMA from the live buffer.
__global__ __launch_bounds__(128) void gemm_bf16_kernel(
    const bf16* __restrict__ A, int lda,
    const float* __restrict__ Bw, int ldb, int head_cols, long long head_stride,
    const float* __restrict__ bias,
    float* __restrict__ Cf, bf16* __restrict__ Cb, int ldc,
    const float* __restrict__ Res,
    int N, int K, int act) {
  __shared__ __align__(64) bf16 BT[2][128][32];   // weight tiles, [N][K] bf16
  const int tid  = threadIdx.x;
  const unsigned lane = tid & 31u;
  const unsigned hi   = lane >> 4;
  const int wave = tid >> 5;
  const int wm = wave >> 1, wn = wave & 1;
  const int nbase = blockIdx.x * 128;
  const int mbase = blockIdx.y * 128;

  const v8f z8 = {0.f,0.f,0.f,0.f,0.f,0.f,0.f,0.f};
  v8f acc[4][4];
#pragma unroll
  for (int i = 0; i < 4; ++i)
#pragma unroll
    for (int j = 0; j < 4; ++j) acc[i][j] = z8;

  // staging pointer for this thread's weight column
  const int ng = nbase + tid;
  const float* bp = nullptr;
  if (ng < N) {
    long long head = ng / head_cols;
    long long ncol = ng % head_cols;
    bp = Bw + head * head_stride + ncol;
  }

  // prologue: fetch tile 0 into registers (coalesced across the 128 threads)
  float r[32];
#pragma unroll
  for (int k = 0; k < 32; ++k) r[k] = bp ? bp[(size_t)k * ldb] : 0.f;

  int buf = 0;
  for (int k0 = 0; k0 < K; k0 += 32) {
    // convert + pack -> 4x b128 LDS stores
    union { bf16 h[32]; u32x4 q[4]; } u;
#pragma unroll
    for (int k = 0; k < 32; ++k) u.h[k] = (bf16)r[k];
#pragma unroll
    for (int qq = 0; qq < 4; ++qq)
      *(u32x4*)&BT[buf][tid][qq * 8] = u.q[qq];
    __syncthreads();

    // kick off next tile's global loads while this tile computes
    if (k0 + 32 < K) {
      if (bp) {
        if (k0 + 64 < K)
          __builtin_prefetch(bp + (size_t)(k0 + 64) * ldb, 0, 1);
#pragma unroll
        for (int k = 0; k < 32; ++k) r[k] = bp[(size_t)(k0 + 32 + k) * ldb];
      }
    }

    v16bf af[4], bfr[4];
#pragma unroll
    for (int mt = 0; mt < 4; ++mt)
      af[mt] = load_a_frag(A + (size_t)(mbase + wm*64 + mt*16 + (lane & 15)) * lda + k0);
#pragma unroll
    for (int nt = 0; nt < 4; ++nt)
      bfr[nt] = load_b_frag(&BT[buf][wn*64 + nt*16 + (lane & 15)][0]);
#pragma unroll
    for (int mt = 0; mt < 4; ++mt)
#pragma unroll
      for (int nt = 0; nt < 4; ++nt)
        acc[mt][nt] = wmma_bf16(af[mt], bfr[nt], acc[mt][nt]);
    buf ^= 1;
  }

  // epilogue: C layout VGPR j -> (M = j + 8*hi, N = lane&15)
#pragma unroll
  for (int mt = 0; mt < 4; ++mt)
#pragma unroll
    for (int nt = 0; nt < 4; ++nt)
#pragma unroll
      for (int j = 0; j < 8; ++j) {
        int row = mbase + wm*64 + mt*16 + j + (hi ? 8 : 0);
        int col = nbase + wn*64 + nt*16 + (int)(lane & 15);
        if (col < N) {
          float v = acc[mt][nt][j] + bias[col];
          if (Res) v += Res[(size_t)row * ldc + col];
          if (act == 1) v = gelu_f(v);
          if (Cf) Cf[(size_t)row * ldc + col] = v;
          if (Cb) Cb[(size_t)row * ldc + col] = (bf16)v;
        }
      }
}

// ---------------- per-head low-rank V up-projection ----------------
// v[m, h*64+e] = sum_r vd[m, h*64+r] * Wvu[h][r][e] + bvu[h][e]
__global__ __launch_bounds__(128) void vu_kernel(const bf16* __restrict__ vdb,
                                                 const float* __restrict__ Wvu,
                                                 const float* __restrict__ bvu,
                                                 bf16* __restrict__ vb) {
  __shared__ __align__(64) bf16 WT[64][64];  // WT[e][r]
  const int h = blockIdx.y;
  const int mbase = blockIdx.x * 64;
  const int tid = threadIdx.x;
  const unsigned lane = tid & 31u;
  const unsigned hi   = lane >> 4;
  const int wave = tid >> 5;

  const float* W = Wvu + (size_t)h * RNK * HD;
  for (int i = tid; i < 64 * 64; i += 128) {
    int r = i >> 6, e = i & 63;
    WT[e][r] = (bf16)W[i];
  }
  __syncthreads();

  const int mrow = mbase + wave * 16 + (int)(lane & 15);
  const bf16* arow = vdb + (size_t)mrow * DIM + h * 64;
  v16bf a0 = load_a_frag(arow);
  v16bf a1 = load_a_frag(arow + 32);

  const v8f z8 = {0.f,0.f,0.f,0.f,0.f,0.f,0.f,0.f};
  v8f o[4] = {z8, z8, z8, z8};
#pragma unroll
  for (int nt = 0; nt < 4; ++nt) {
    const bf16* wr = &WT[nt*16 + (lane & 15)][0];
    o[nt] = wmma_bf16(a0, load_b_frag(wr),      o[nt]);
    o[nt] = wmma_bf16(a1, load_b_frag(wr + 32), o[nt]);
  }
#pragma unroll
  for (int nt = 0; nt < 4; ++nt)
#pragma unroll
    for (int j = 0; j < 8; ++j) {
      int row = mbase + wave*16 + j + (hi ? 8 : 0);
      int e   = nt*16 + (int)(lane & 15);
      vb[(size_t)row * DIM + h*64 + e] = (bf16)(o[nt][j] + bvu[h*64 + e]);
    }
}

// ---------------- flash attention (mask keeps t >= s) ----------------
// Block: 64 query rows of one (b,h); 4 waves x 16 rows. Streams 64-key blocks
// with a double-buffered V^T tile (one barrier per key block).
__global__ __launch_bounds__(128) void attn_kernel(const bf16* __restrict__ qb,
                                                   const bf16* __restrict__ kb,
                                                   const bf16* __restrict__ vb,
                                                   float* __restrict__ x) {
  __shared__ __align__(64) bf16 VT[2][64][64];    // V^T for current key block
  __shared__ __align__(64) bf16 Pt[4][16][64];    // per-wave P tile (C->A bounce)
  const int qblock = blockIdx.x * 64;
  const int h = blockIdx.y, b = blockIdx.z;
  const int tid = threadIdx.x;
  const unsigned lane = tid & 31u;
  const unsigned hi   = lane >> 4;
  const int wave = tid >> 5;
  const int qw = qblock + wave * 16;
  const float scale = 0.03125f;                   // 1/sqrt(DIM)

  const bf16* qrow = qb + (size_t)(b*SEQ + qw + (int)(lane & 15)) * DIM + h * 64;
  v16bf qa0 = load_a_frag(qrow);
  v16bf qa1 = load_a_frag(qrow + 32);

  const v8f z8 = {0.f,0.f,0.f,0.f,0.f,0.f,0.f,0.f};
  v8f o[4] = {z8, z8, z8, z8};
  float mrun[8], lrun[8];
#pragma unroll
  for (int j = 0; j < 8; ++j) { mrun[j] = -1e30f; lrun[j] = 0.f; }

  int buf = 0;
  for (int t0 = qblock; t0 < SEQ; t0 += 64) {     // only t >= s is kept
    // stage V^T: 2 threads per key row, 32 e each (coalesced in e)
    {
      int tr = tid >> 1, e0 = (tid & 1) * 32;
      const bf16* vrow = vb + (size_t)(b*SEQ + t0 + tr) * DIM + h*64 + e0;
#pragma unroll
      for (int i = 0; i < 32; ++i) VT[buf][e0 + i][tr] = vrow[i];
    }
    __syncthreads();

    // S = Q K^T : B-fragments straight from global (K rows contiguous in e)
    v8f s[4] = {z8, z8, z8, z8};
#pragma unroll
    for (int nt = 0; nt < 4; ++nt) {
      const bf16* krow = kb + (size_t)(b*SEQ + t0 + nt*16 + (int)(lane & 15)) * DIM + h*64;
      s[nt] = wmma_bf16(qa0, load_b_frag(krow),      s[nt]);
      s[nt] = wmma_bf16(qa1, load_b_frag(krow + 32), s[nt]);
    }

    // mask + online softmax update (row stats live in 16-lane halves)
#pragma unroll
    for (int j = 0; j < 8; ++j) {
      int srow = qw + j + (hi ? 8 : 0);
      float mj = -1e30f;
#pragma unroll
      for (int nt = 0; nt < 4; ++nt) {
        int t = t0 + nt*16 + (int)(lane & 15);
        float v = s[nt][j] * scale;
        v = (t >= srow) ? v : -1e30f;
        s[nt][j] = v;
        mj = fmaxf(mj, v);
      }
      mj = fmaxf(mj, __shfl_xor(mj, 1));
      mj = fmaxf(mj, __shfl_xor(mj, 2));
      mj = fmaxf(mj, __shfl_xor(mj, 4));
      mj = fmaxf(mj, __shfl_xor(mj, 8));
      float mnew = fmaxf(mrun[j], mj);
      float sc   = __expf(mrun[j] - mnew);
      mrun[j] = mnew;
      float rs = 0.f;
#pragma unroll
      for (int nt = 0; nt < 4; ++nt) {
        float p = __expf(s[nt][j] - mnew);
        s[nt][j] = p;
        rs += p;
      }
      rs += __shfl_xor(rs, 1);
      rs += __shfl_xor(rs, 2);
      rs += __shfl_xor(rs, 4);
      rs += __shfl_xor(rs, 8);
      lrun[j] = lrun[j] * sc + rs;
      o[0][j] *= sc; o[1][j] *= sc; o[2][j] *= sc; o[3][j] *= sc;
#pragma unroll
      for (int nt = 0; nt < 4; ++nt)
        Pt[wave][j + (hi ? 8 : 0)][nt*16 + (lane & 15)] = (bf16)s[nt][j];
    }
    // same-wave LDS RAW: drain DS counter before re-reading P as A-fragments
    asm volatile("s_wait_dscnt 0x0" ::: "memory");

    const bf16* prow = &Pt[wave][lane & 15][0];
    v16bf pa0 = load_a_frag(prow);
    v16bf pa1 = load_a_frag(prow + 32);
#pragma unroll
    for (int nt = 0; nt < 4; ++nt) {
      const bf16* vr = &VT[buf][nt*16 + (lane & 15)][0];
      o[nt] = wmma_bf16(pa0, load_b_frag(vr),      o[nt]);
      o[nt] = wmma_bf16(pa1, load_b_frag(vr + 32), o[nt]);
    }
    buf ^= 1;
  }

  // x += O / l   (disjoint (row,col) per thread across the whole grid)
#pragma unroll
  for (int nt = 0; nt < 4; ++nt)
#pragma unroll
    for (int j = 0; j < 8; ++j) {
      int row = qw + j + (hi ? 8 : 0);
      int col = h*64 + nt*16 + (int)(lane & 15);
      x[(size_t)(b*SEQ + row) * DIM + col] += o[nt][j] / lrun[j];
    }
}

// ---------------- loss ----------------
__global__ void zero2_kernel(float* acc) { acc[0] = 0.f; acc[1] = 0.f; }

__global__ __launch_bounds__(256) void loss_kernel(const float* __restrict__ logits,
                                                   const int* __restrict__ labels,
                                                   float* __restrict__ acc) {
  __shared__ float red[256];
  int row = blockIdx.x, tid = threadIdx.x;
  const float* lr = logits + (size_t)row * VOC;
  float m = -1e30f;
  for (int i = tid; i < VOC; i += 256) m = fmaxf(m, lr[i]);
  red[tid] = m; __syncthreads();
  for (int off = 128; off > 0; off >>= 1) {
    if (tid < off) red[tid] = fmaxf(red[tid], red[tid + off]);
    __syncthreads();
  }
  m = red[0]; __syncthreads();
  float s = 0.f;
  for (int i = tid; i < VOC; i += 256) s += __expf(lr[i] - m);
  red[tid] = s; __syncthreads();
  for (int off = 128; off > 0; off >>= 1) {
    if (tid < off) red[tid] += red[tid + off];
    __syncthreads();
  }
  if (tid == 0) {
    float lse = logf(red[0]) + m;
    int lbl = labels[row];
    int cl  = lbl < 0 ? 0 : lbl;
    float nll = lse - lr[cl];
    if (lbl != -1) {
      atomicAdd(acc, nll);
      atomicAdd(acc + 1, 1.0f);
    }
  }
}

__global__ void loss_fin_kernel(const float* acc, float* out) {
  out[0] = acc[0] / acc[1];
}

// ---------------- host driver ----------------
extern "C" void kernel_launch(void* const* d_in, const int* in_sizes, int n_in,
                              void* d_out, int out_size, void* d_ws, size_t ws_size,
                              hipStream_t stream) {
  const int*   idx_seq = (const int*)d_in[0];
  const int*   labels  = (const int*)d_in[1];
  const float* wte  = (const float*)d_in[2];
  const float* wpe  = (const float*)d_in[3];
  const float* ln1g = (const float*)d_in[4];
  const float* ln1b = (const float*)d_in[5];
  const float* Wq   = (const float*)d_in[6];
  const float* bq   = (const float*)d_in[7];
  const float* Wk   = (const float*)d_in[8];
  const float* bk   = (const float*)d_in[9];
  const float* Wvd  = (const float*)d_in[10];
  const float* bvd  = (const float*)d_in[11];
  const float* Wvu  = (const float*)d_in[12];
  const float* bvu  = (const float*)d_in[13];
  const float* ln2g = (const float*)d_in[14];
  const float* ln2b = (const float*)d_in[15];
  const float* W1   = (const float*)d_in[16];
  const float* b1   = (const float*)d_in[17];
  const float* W2   = (const float*)d_in[18];
  const float* b2   = (const float*)d_in[19];
  const float* lnfg = (const float*)d_in[20];
  const float* lnfb = (const float*)d_in[21];
  const float* Wh   = (const float*)d_in[22];
  const float* bh   = (const float*)d_in[23];
  float* out = (float*)d_out;

  // workspace carve-up (~44 MB)
  char* ws = (char*)d_ws;
  size_t off = 0;
  auto carve = [&](size_t bytes) { void* p = ws + off; off += (bytes + 255) & ~(size_t)255; return p; };
  float* x    = (float*)carve((size_t)MROWS * DIM * 4);
  bf16*  hb   = (bf16*) carve((size_t)MROWS * DIM * 2);
  bf16*  qbuf = (bf16*) carve((size_t)MROWS * DIM * 2);
  bf16*  kbuf = (bf16*) carve((size_t)MROWS * DIM * 2);
  bf16*  vdb  = (bf16*) carve((size_t)MROWS * DIM * 2);
  bf16*  vbuf = (bf16*) carve((size_t)MROWS * DIM * 2);
  bf16*  m1b  = (bf16*) carve((size_t)MROWS * FF  * 2);
  float* acc  = (float*)carve(256);

  const dim3 blk128(128), blk256(256);
  const dim3 gN1024(DIM / 128, MROWS / 128);          // (8,16)
  const dim3 gFF(FF / 128, MROWS / 128);              // (32,16)
  const dim3 gVOC((VOC + 127) / 128, MROWS / 128);    // (393,16)

  embed_kernel<<<MROWS, blk256, 0, stream>>>(idx_seq, wte, wpe, x);

  for (int l = 0; l < LNUM; ++l) {
    const float* Wq_l  = Wq  + (size_t)l * HN * DIM * HD;
    const float* Wk_l  = Wk  + (size_t)l * HN * DIM * HD;
    const float* Wvd_l = Wvd + (size_t)l * HN * DIM * RNK;
    const float* Wvu_l = Wvu + (size_t)l * HN * RNK * HD;
    const float* W1_l  = W1  + (size_t)l * DIM * FF;
    const float* W2_l  = W2  + (size_t)l * FF * DIM;

    ln_kernel<<<MROWS, blk256, 0, stream>>>(x, ln1g + l*DIM, ln1b + l*DIM, hb);

    // Q = hb @ Wq[l] + bq  (per-head weight layout [H,D,64])
    gemm_bf16_kernel<<<gN1024, blk128, 0, stream>>>(
        hb, DIM, Wq_l, HD, 64, (long long)DIM * HD, bq + (size_t)l*HN*HD,
        nullptr, qbuf, DIM, nullptr, DIM, DIM, 0);
    gemm_bf16_kernel<<<gN1024, blk128, 0, stream>>>(
        hb, DIM, Wk_l, HD, 64, (long long)DIM * HD, bk + (size_t)l*HN*HD,
        nullptr, kbuf, DIM, nullptr, DIM, DIM, 0);
    gemm_bf16_kernel<<<gN1024, blk128, 0, stream>>>(
        hb, DIM, Wvd_l, RNK, 64, (long long)DIM * RNK, bvd + (size_t)l*HN*RNK,
        nullptr, vdb, DIM, nullptr, DIM, DIM, 0);

    vu_kernel<<<dim3(MROWS / 64, HN), blk128, 0, stream>>>(
        vdb, Wvu_l, bvu + (size_t)l*HN*HD, vbuf);

    attn_kernel<<<dim3(SEQ / 64, HN, BAT), blk128, 0, stream>>>(qbuf, kbuf, vbuf, x);

    ln_kernel<<<MROWS, blk256, 0, stream>>>(x, ln2g + l*DIM, ln2b + l*DIM, hb);

    // MLP1 with fused GELU -> bf16
    gemm_bf16_kernel<<<gFF, blk128, 0, stream>>>(
        hb, DIM, W1_l, FF, FF, 0LL, b1 + (size_t)l*FF,
        nullptr, m1b, FF, nullptr, FF, DIM, 1);
    // MLP2 with fused residual add, in-place into x
    gemm_bf16_kernel<<<gN1024, blk128, 0, stream>>>(
        m1b, FF, W2_l, DIM, DIM, 0LL, b2 + (size_t)l*DIM,
        x, nullptr, DIM, x, DIM, FF, 0);
  }

  ln_kernel<<<MROWS, blk256, 0, stream>>>(x, lnfg, lnfb, hb);

  // LM head -> logits (f32) straight into d_out
  gemm_bf16_kernel<<<gVOC, blk128, 0, stream>>>(
      hb, DIM, Wh, VOC, VOC, 0LL, bh,
      out, nullptr, VOC, nullptr, VOC, DIM, 0);

  zero2_kernel<<<1, 1, 0, stream>>>(acc);
  loss_kernel<<<MROWS, blk256, 0, stream>>>(out, labels, acc);
  loss_fin_kernel<<<1, 1, 0, stream>>>(acc, out + (size_t)MROWS * VOC);
}
